// CustomFourierLayer_26182120636504
// MI455X (gfx1250) — matrix operand order repaired and checked by reading, not compiled, not run on hardware
//
#include <hip/hip_runtime.h>

typedef __attribute__((ext_vector_type(16))) _Float16 v16h;
typedef __attribute__((ext_vector_type(8)))  float    v8f;
typedef __attribute__((ext_vector_type(4)))  float    f32x4;

#define BDIM   16384
#define IDIM   512
#define ODIM   64
#define KTOT   4128      // 4096 real features + 32-wide const-padding chunk
#define WAVES  8         // waves per block
#define ROWS_PER_BLOCK (WAVES * 16)

// ---------------------------------------------------------------------------
// Prep kernel 1: fold weights into coefficients, store column-major f16
// Wc[o][k] = weights[o][i] * coef[o][i][1+c],  k = i*8 + c, c in 0..7
// feature order per i: sin1,cos1,sin2,cos2,sin3,cos3,sin4,cos4
// ---------------------------------------------------------------------------
__global__ __launch_bounds__(256) void prep_wc(const float* __restrict__ w,
                                               const float* __restrict__ coef,
                                               _Float16* __restrict__ wc) {
    int idx = blockIdx.x * 256 + threadIdx.x;   // 0 .. 64*4096-1
    int n = idx >> 12;
    int k = idx & 4095;
    int i = k >> 3;
    int c = k & 7;
    float v = w[n * IDIM + i] * coef[(n * IDIM + i) * 9 + 1 + c];
    wc[n * KTOT + k] = (_Float16)v;
}

// ---------------------------------------------------------------------------
// Prep kernel 2: const term as feature 4096 (value 1.0), zero the pad.
// Wc[o][4096] = sum_i w[o][i]*coef[o][i][0];  Wc[o][4097..4127] = 0
// ---------------------------------------------------------------------------
__global__ __launch_bounds__(32) void prep_const(const float* __restrict__ w,
                                                 const float* __restrict__ coef,
                                                 _Float16* __restrict__ wc) {
    int n = blockIdx.x;
    int lane = threadIdx.x;           // 0..31 (wave32)
    float s = 0.f;
    for (int j = lane; j < IDIM; j += 32)
        s += w[n * IDIM + j] * coef[(n * IDIM + j) * 9];
    for (int off = 16; off > 0; off >>= 1)
        s += __shfl_down(s, off, 32);
    s = __shfl(s, 0, 32);
    wc[n * KTOT + 4096 + lane] = (lane == 0) ? (_Float16)s : (_Float16)0.f;
}

// ---------------------------------------------------------------------------
// Eight Fourier features of one x value via sincos + angle-addition.
// ---------------------------------------------------------------------------
__device__ __forceinline__ void feats(float xv, _Float16* e) {
    float s1, c1;
    __sincosf(xv, &s1, &c1);
    float s2 = 2.f * s1 * c1;
    float c2 = c1 * c1 - s1 * s1;
    float s3 = s2 * c1 + c2 * s1;
    float c3 = c2 * c1 - s2 * s1;
    float s4 = s3 * c1 + c3 * s1;
    float c4 = c3 * c1 - s3 * s1;
    e[0] = (_Float16)s1; e[1] = (_Float16)c1;
    e[2] = (_Float16)s2; e[3] = (_Float16)c2;
    e[4] = (_Float16)s3; e[5] = (_Float16)c3;
    e[6] = (_Float16)s4; e[7] = (_Float16)c4;
}

union AFrag { v16h v; _Float16 e[16]; };
union BFrag { v16h v; f32x4 f[2]; };

__device__ __forceinline__ v16h loadB(const _Float16* p) {
    BFrag b;
    b.f[0] = *(const f32x4*)(p);
    b.f[1] = *(const f32x4*)(p + 8);
    return b.v;
}

// ---------------------------------------------------------------------------
// Main kernel: out[16384,64] = Features(x) @ Wc   (f16 WMMA, f32 accum)
// One wave = 16 rows x 64 outputs; 129 k-steps of 32 features.
// ---------------------------------------------------------------------------
__global__ __launch_bounds__(256) void fourier_wmma(const float* __restrict__ x,
                                                    const _Float16* __restrict__ wc,
                                                    float* __restrict__ out) {
    __shared__ float xt[WAVES][16][68];   // 68-float row stride: conflict-free, 16B aligned

    const int tid   = threadIdx.x;
    const int wave  = tid >> 5;
    const int lane  = tid & 31;
    const int row16 = lane & 15;
    const int hi    = lane >> 4;          // 0: K 0..15, 1: K 16..31
    const int rowbase = (blockIdx.x * WAVES + wave) * 16;

    v8f acc0 = {}, acc1 = {}, acc2 = {}, acc3 = {};

    // B-fragment base for this lane: column-major Wc, 16 contiguous K per lane
    const _Float16* wbase = wc + row16 * KTOT + hi * 16;

    for (int cc = 0; cc < 8; ++cc) {      // 8 column chunks of 64 x-columns
        // ---- stage 16 rows x 64 cols of x into LDS (coalesced 16B loads) ----
        #pragma unroll
        for (int it = 0; it < 8; ++it) {
            int idx = it * 32 + lane;
            int r   = idx >> 4;
            int c4  = idx & 15;
            f32x4 v = *(const f32x4*)(x + (rowbase + r) * IDIM + cc * 64 + c4 * 4);
            *(f32x4*)&xt[wave][r][c4 * 4] = v;
        }
        if (cc < 7) {   // hint next chunk into cache (global_prefetch_b8)
            __builtin_prefetch(x + (rowbase + (lane >> 1)) * IDIM + (cc + 1) * 64 + (lane & 1) * 32, 0, 0);
        }
        __syncthreads();

        // ---- 16 k-steps of 32 features (4 x-columns each) ----
        #pragma unroll 4
        for (int t = 0; t < 16; ++t) {
            float xa = xt[wave][row16][4 * t + 2 * hi];
            float xb = xt[wave][row16][4 * t + 2 * hi + 1];
            AFrag A;
            feats(xa, A.e);
            feats(xb, A.e + 8);

            const _Float16* p = wbase + cc * 512 + t * 32;
            v16h b0 = loadB(p);
            v16h b1 = loadB(p + 16 * KTOT);
            v16h b2 = loadB(p + 32 * KTOT);
            v16h b3 = loadB(p + 48 * KTOT);

            acc0 = __builtin_amdgcn_wmma_f32_16x16x32_f16(false, A.v, false, b0, (short)0, acc0, false, false);
            acc1 = __builtin_amdgcn_wmma_f32_16x16x32_f16(false, A.v, false, b1, (short)0, acc1, false, false);
            acc2 = __builtin_amdgcn_wmma_f32_16x16x32_f16(false, A.v, false, b2, (short)0, acc2, false, false);
            acc3 = __builtin_amdgcn_wmma_f32_16x16x32_f16(false, A.v, false, b3, (short)0, acc3, false, false);
        }
        __syncthreads();
    }

    // ---- constant chunk (feature 4096 == 1.0, rest of chunk zero) ----
    {
        AFrag A;
        #pragma unroll
        for (int e = 0; e < 16; ++e) A.e[e] = (_Float16)0.f;
        A.e[0] = hi ? (_Float16)0.f : (_Float16)1.0f;   // lanes 0-15 carry K=0 -> feature 4096

        const _Float16* p = wbase + 4096;
        v16h b0 = loadB(p);
        v16h b1 = loadB(p + 16 * KTOT);
        v16h b2 = loadB(p + 32 * KTOT);
        v16h b3 = loadB(p + 48 * KTOT);
        acc0 = __builtin_amdgcn_wmma_f32_16x16x32_f16(false, A.v, false, b0, (short)0, acc0, false, false);
        acc1 = __builtin_amdgcn_wmma_f32_16x16x32_f16(false, A.v, false, b1, (short)0, acc1, false, false);
        acc2 = __builtin_amdgcn_wmma_f32_16x16x32_f16(false, A.v, false, b2, (short)0, acc2, false, false);
        acc3 = __builtin_amdgcn_wmma_f32_16x16x32_f16(false, A.v, false, b3, (short)0, acc3, false, false);
    }

    // ---- write back: VGPR j, lanes 0-15 -> row j; lanes 16-31 -> row j+8 ----
    #pragma unroll
    for (int j = 0; j < 8; ++j) {
        float* orow = out + (rowbase + hi * 8 + j) * ODIM + row16;
        orow[0]  = acc0[j];
        orow[16] = acc1[j];
        orow[32] = acc2[j];
        orow[48] = acc3[j];
    }
}

// ---------------------------------------------------------------------------
extern "C" void kernel_launch(void* const* d_in, const int* in_sizes, int n_in,
                              void* d_out, int out_size, void* d_ws, size_t ws_size,
                              hipStream_t stream) {
    const float* x    = (const float*)d_in[0];   // (16384, 512)
    const float* w    = (const float*)d_in[1];   // (64, 512)
    const float* coef = (const float*)d_in[2];   // (64, 512, 9)
    float* out        = (float*)d_out;           // (16384, 64)
    _Float16* wc      = (_Float16*)d_ws;         // 64 * 4128 f16 = 516 KiB

    prep_wc<<<(ODIM * 4096) / 256, 256, 0, stream>>>(w, coef, wc);
    prep_const<<<ODIM, 32, 0, stream>>>(w, coef, wc);
    fourier_wmma<<<BDIM / ROWS_PER_BLOCK, 256, 0, stream>>>(x, wc, out);
}
